// TransformerBlock_87754771792205
// MI455X (gfx1250) — matrix operand order repaired and checked by reading, not compiled, not run on hardware
//
#include <hip/hip_runtime.h>
#include <hip/hip_bf16.h>
#include <math.h>

// ---------------------------------------------------------------------------
// Transformer block (pre-LN, ViT style) for x:[1,256,64,64], heads=8, d=32.
// All GEMMs + attention run on CDNA5 WMMA (v_wmma_f32_16x16x32_f16, wave32).
// GEMM: 32x32 tile per wave (4 WMMAs / K-step), 128x64 block tile.
// Attention: flash, transposed (S^T = K*Q^T, O^T = V^T*P^T), block-shared
// K/V LDS staging.
// ---------------------------------------------------------------------------

typedef _Float16 h16 __attribute__((ext_vector_type(16)));
typedef _Float16 h8v __attribute__((ext_vector_type(8)));
typedef float    f8v __attribute__((ext_vector_type(8)));

#define TOK   4096
#define CDIM  256
#define HEADS 8
#define HDIM  32

// ------------------------------- fp32 -> fp16 ------------------------------
__global__ void cvt_f16_kernel(const float* __restrict__ in,
                               _Float16* __restrict__ out, int n) {
  int i = blockIdx.x * 256 + threadIdx.x;
  if (i < n) out[i] = (_Float16)in[i];
}

// ----------------------------- LayerNorm -----------------------------------
// one wave per token; chanFirst=1 reads x[c*4096+n] (NCHW input), else [n*256+c]
__global__ __launch_bounds__(256)
void ln_kernel(const float* __restrict__ src, const float* __restrict__ g,
               const float* __restrict__ b, float* __restrict__ xl_out,
               _Float16* __restrict__ xn_out, int chanFirst) {
  const int lane = threadIdx.x & 31;
  const int wave = threadIdx.x >> 5;
  const int n = blockIdx.x * 8 + wave;
  float v[8];
  float s = 0.f, sq = 0.f;
#pragma unroll
  for (int i = 0; i < 8; ++i) {
    int c = lane + 32 * i;
    float xv = chanFirst ? src[(size_t)c * TOK + n] : src[(size_t)n * CDIM + c];
    v[i] = xv; s += xv; sq += xv * xv;
  }
#pragma unroll
  for (int m = 1; m < 32; m <<= 1) {
    s  += __shfl_xor(s,  m, 32);
    sq += __shfl_xor(sq, m, 32);
  }
  float mu  = s * (1.f / CDIM);
  float var = sq * (1.f / CDIM) - mu * mu;
  float rs  = rsqrtf(var + 1e-5f);
#pragma unroll
  for (int i = 0; i < 8; ++i) {
    int c = lane + 32 * i;
    if (xl_out) xl_out[(size_t)n * CDIM + c] = v[i];
    xn_out[(size_t)n * CDIM + c] = (_Float16)((v[i] - mu) * rs * g[c] + b[c]);
  }
}

// ------------------------------ WMMA GEMM ----------------------------------
// C[m,n] = sum_k A[m,k] * W[n,k]  (A:[M,K] f16 row-major, W:[N,K] f16 row-major)
// 32x32 tile per wave; block tile 128(M) x 64(N); 8 waves (4 M x 2 N).
// mode 0: +bias            -> f16 out [M,N]
// mode 1: +bias +res f32   -> f32 out [M,N]
// mode 2: +bias, GELU      -> f16 out [M,N]
// mode 3: +bias +res f32   -> f32 out TRANSPOSED: out[n*4096 + m]
__global__ __launch_bounds__(256)
void gemm_wmma_kernel(const _Float16* __restrict__ A,
                      const _Float16* __restrict__ W,
                      const float* __restrict__ bias,
                      const float* __restrict__ res,
                      void* __restrict__ out,
                      int M, int N, int K, int mode) {
  const int lane = threadIdx.x & 31;
  const int wave = threadIdx.x >> 5;
  const int lo = lane & 15;
  const int hi = lane >> 4;
  const int m0 = blockIdx.y * 128 + (wave & 3) * 32;
  const int n0 = blockIdx.x * 64 + (wave >> 2) * 32;

  const _Float16* ap0 = A + (size_t)(m0 + lo) * K + hi * 8;        // rows m0..+15
  const _Float16* ap1 = ap0 + (size_t)16 * K;                      // rows m0+16..+31
  const _Float16* bp0 = W + (size_t)(n0 + lo) * K + hi * 8;        // cols n0..+15
  const _Float16* bp1 = bp0 + (size_t)16 * K;                      // cols n0+16..+31

  f8v c00 = {}, c01 = {}, c10 = {}, c11 = {};
  for (int k = 0; k < K; k += 32) {
    __builtin_prefetch(ap0 + 128, 0, 3);
    __builtin_prefetch(bp0 + 128, 0, 3);
    h8v a00 = *(const h8v*)(ap0);
    h8v a01 = *(const h8v*)(ap0 + 16);
    h8v a10 = *(const h8v*)(ap1);
    h8v a11 = *(const h8v*)(ap1 + 16);
    h8v b00 = *(const h8v*)(bp0);
    h8v b01 = *(const h8v*)(bp0 + 16);
    h8v b10 = *(const h8v*)(bp1);
    h8v b11 = *(const h8v*)(bp1 + 16);
    ap0 += 32; ap1 += 32; bp0 += 32; bp1 += 32;
    h16 a0, a1, b0, b1;
#pragma unroll
    for (int i = 0; i < 8; ++i) {
      a0[i] = a00[i]; a0[i + 8] = a01[i];
      a1[i] = a10[i]; a1[i + 8] = a11[i];
      b0[i] = b00[i]; b0[i + 8] = b01[i];
      b1[i] = b10[i]; b1[i + 8] = b11[i];
    }
    c00 = __builtin_amdgcn_wmma_f32_16x16x32_f16(false, a0, false, b0, (short)0, c00, false, false);
    c01 = __builtin_amdgcn_wmma_f32_16x16x32_f16(false, a0, false, b1, (short)0, c01, false, false);
    c10 = __builtin_amdgcn_wmma_f32_16x16x32_f16(false, a1, false, b0, (short)0, c10, false, false);
    c11 = __builtin_amdgcn_wmma_f32_16x16x32_f16(false, a1, false, b1, (short)0, c11, false, false);
  }

  f8v accs[2][2] = {{c00, c01}, {c10, c11}};
#pragma unroll
  for (int ti = 0; ti < 2; ++ti) {
#pragma unroll
    for (int tj = 0; tj < 2; ++tj) {
      const int n = n0 + tj * 16 + lo;
      const float bv = bias[n];
#pragma unroll
      for (int j = 0; j < 8; ++j) {
        int m = m0 + ti * 16 + j + 8 * hi;
        float v = accs[ti][tj][j] + bv;
        if (mode == 0) {
          ((_Float16*)out)[(size_t)m * N + n] = (_Float16)v;
        } else if (mode == 1) {
          v += res[(size_t)m * N + n];
          ((float*)out)[(size_t)m * N + n] = v;
        } else if (mode == 2) {
          float gv = 0.5f * v * (1.0f + erff(v * 0.70710678118654752f));
          ((_Float16*)out)[(size_t)m * N + n] = (_Float16)gv;
        } else {  // mode 3: final residual + NHWC->NCHW transpose store
          v += res[(size_t)m * CDIM + n];
          ((float*)out)[(size_t)n * TOK + m] = v;
        }
      }
    }
  }
}

// --------------------------- Flash attention -------------------------------
// qkv:[4096,768] f16 (q|k|v each 256 cols). One 16-query tile per wave;
// all 8 waves share one head and one block-wide K/V LDS tile per 32-key chunk.
// Computes S^T = K*Q^T and O^T = V^T*P^T so softmax stats are per-lane uniform.
__global__ __launch_bounds__(256)
void attn_kernel(const _Float16* __restrict__ qkv, _Float16* __restrict__ out) {
  __shared__ __attribute__((aligned(16))) _Float16 ldsK[32][40];  // 80B rows
  __shared__ __attribute__((aligned(16))) _Float16 ldsV[32][40];
  const int lane = threadIdx.x & 31;
  const int wave = threadIdx.x >> 5;
  const int lo = lane & 15;
  const int hi = lane >> 4;
  const int h = blockIdx.y;
  const int qbase = (blockIdx.x * 8 + wave) * 16;

  // staging assignment: 256 threads cover 2 matrices x 32 rows x 4 x 16B
  const int sKV  = threadIdx.x >> 7;         // 0 = K, 1 = V
  const int sRow = (threadIdx.x & 127) >> 2; // 0..31
  const int sQrt = threadIdx.x & 3;          // 16B quarter of a 64B row

  // Q as B-operand of S^T: lane holds query row qbase+lo, d = hi*8..+7 and 16+hi*8..+7
  h16 qf;
  {
    const _Float16* qp = qkv + (size_t)(qbase + lo) * 768 + h * HDIM + hi * 8;
    h8v q0 = *(const h8v*)qp;
    h8v q1 = *(const h8v*)(qp + 16);
#pragma unroll
    for (int i = 0; i < 8; ++i) { qf[i] = q0[i]; qf[i + 8] = q1[i]; }
  }
  const _Float16* KVb = qkv + CDIM + (size_t)sKV * CDIM + h * HDIM;  // per staging role

  f8v o0 = {}, o1 = {};
  float mrun = -1e30f, lrun = 0.f;
  const float scale = 0.17677669529663689f;  // 1/sqrt(32)

  for (int kb = 0; kb < TOK; kb += 32) {
    // cooperative block-wide stage of K and V chunk (16B per thread)
    {
      const _Float16* src = KVb + (size_t)(kb + sRow) * 768 + sQrt * 8;
      _Float16* dst = (sKV ? &ldsV[sRow][sQrt * 8] : &ldsK[sRow][sQrt * 8]);
      *(h8v*)dst = *(const h8v*)src;
    }
    __syncthreads();  // uniform trip count across all waves

    // S^T tiles: rows = keys, cols = queries. A = K rows (from LDS), B = qf.
    f8v st[2];
#pragma unroll
    for (int t = 0; t < 2; ++t) {
      const _Float16* kp = &ldsK[t * 16 + lo][hi * 8];
      h8v k0 = *(const h8v*)kp;
      h8v k1 = *(const h8v*)(kp + 16);
      h16 kf;
#pragma unroll
      for (int i = 0; i < 8; ++i) { kf[i] = k0[i]; kf[i + 8] = k1[i]; }
      f8v z = {};
      st[t] = __builtin_amdgcn_wmma_f32_16x16x32_f16(false, kf, false, qf,
                                                     (short)0, z, false, false);
    }

    // online softmax over this 32-key chunk (per query = per lane pair)
    float cmax = -1e30f;
#pragma unroll
    for (int j = 0; j < 8; ++j) {
      st[0][j] *= scale; st[1][j] *= scale;
      cmax = fmaxf(cmax, fmaxf(st[0][j], st[1][j]));
    }
    cmax = fmaxf(cmax, __shfl_xor(cmax, 16, 32));
    float mnew  = fmaxf(mrun, cmax);
    float alpha = __expf(mrun - mnew);
    mrun = mnew;
    float ls = 0.f;
#pragma unroll
    for (int j = 0; j < 8; ++j) {
      st[0][j] = __expf(st[0][j] - mnew);
      st[1][j] = __expf(st[1][j] - mnew);
      ls += st[0][j] + st[1][j];
    }
    ls += __shfl_xor(ls, 16, 32);
    lrun = lrun * alpha + ls;
#pragma unroll
    for (int j = 0; j < 8; ++j) { o0[j] *= alpha; o1[j] *= alpha; }

    // P^T (C-layout) -> B-layout: one lane-pair exchange per element
    h16 pf;
#pragma unroll
    for (int j = 0; j < 8; ++j) {
      float a0 = st[0][j], a1 = st[1][j];
      float x0 = __shfl_xor(a0, 16, 32);
      float x1 = __shfl_xor(a1, 16, 32);
      pf[j]     = (_Float16)(hi ? x1 : a0);
      pf[j + 8] = (_Float16)(hi ? a1 : x0);
    }

    // O^T += V^T * P^T : A = V^T (transposed gather from LDS), B = pf
#pragma unroll
    for (int t = 0; t < 2; ++t) {
      int d = t * 16 + lo;
      h16 vf;
#pragma unroll
      for (int i = 0; i < 8; ++i) {
        vf[i]     = ldsV[16 * hi + i][d];
        vf[i + 8] = ldsV[16 * hi + 8 + i][d];
      }
      if (t == 0)
        o0 = __builtin_amdgcn_wmma_f32_16x16x32_f16(false, vf, false, pf,
                                                    (short)0, o0, false, false);
      else
        o1 = __builtin_amdgcn_wmma_f32_16x16x32_f16(false, vf, false, pf,
                                                    (short)0, o1, false, false);
    }
    __syncthreads();  // protect shared K/V from next chunk's overwrite
  }

  // normalize and store: lane holds query qbase+lo, d = j+8*hi (+16 for o1)
  float linv = 1.0f / lrun;
  _Float16* op = out + (size_t)(qbase + lo) * CDIM + h * HDIM;
#pragma unroll
  for (int j = 0; j < 8; ++j) {
    op[j + 8 * hi]      = (_Float16)(o0[j] * linv);
    op[16 + j + 8 * hi] = (_Float16)(o1[j] * linv);
  }
}

// ------------------------------- launcher ----------------------------------
extern "C" void kernel_launch(void* const* d_in, const int* in_sizes, int n_in,
                              void* d_out, int out_size, void* d_ws, size_t ws_size,
                              hipStream_t stream) {
  const float* x      = (const float*)d_in[0];
  const float* ln1_g  = (const float*)d_in[1];
  const float* ln1_b  = (const float*)d_in[2];
  const float* w_qkv  = (const float*)d_in[3];
  const float* b_qkv  = (const float*)d_in[4];
  const float* w_proj = (const float*)d_in[5];
  const float* b_proj = (const float*)d_in[6];
  const float* ln2_g  = (const float*)d_in[7];
  const float* ln2_b  = (const float*)d_in[8];
  const float* w_mlp1 = (const float*)d_in[9];
  const float* b_mlp1 = (const float*)d_in[10];
  const float* w_mlp2 = (const float*)d_in[11];
  const float* b_mlp2 = (const float*)d_in[12];

  char* ws = (char*)d_ws;
  float*    xl      = (float*)(ws + 0);          // [4096,256] f32
  float*    xl2     = (float*)(ws + 4194304);    // [4096,256] f32
  _Float16* xn      = (_Float16*)(ws + 8388608); // [4096,256] f16
  _Float16* xn2     = (_Float16*)(ws + 10485760);
  _Float16* qkv16   = (_Float16*)(ws + 12582912);// [4096,768] f16
  _Float16* att16   = (_Float16*)(ws + 18874368);// [4096,256] f16
  _Float16* h1      = (_Float16*)(ws + 20971520);// [4096,1024] f16
  _Float16* wqkv16  = (_Float16*)(ws + 29360128);
  _Float16* wproj16 = (_Float16*)(ws + 29753344);
  _Float16* wmlp116 = (_Float16*)(ws + 29884416);
  _Float16* wmlp216 = (_Float16*)(ws + 30408704);

  // weight conversion fp32 -> fp16
  cvt_f16_kernel<<<(768 * 256 + 255) / 256, 256, 0, stream>>>(w_qkv, wqkv16, 768 * 256);
  cvt_f16_kernel<<<(256 * 256 + 255) / 256, 256, 0, stream>>>(w_proj, wproj16, 256 * 256);
  cvt_f16_kernel<<<(1024 * 256 + 255) / 256, 256, 0, stream>>>(w_mlp1, wmlp116, 1024 * 256);
  cvt_f16_kernel<<<(256 * 1024 + 255) / 256, 256, 0, stream>>>(w_mlp2, wmlp216, 256 * 1024);

  // LN1 (+ NCHW -> [N,C] transpose, keeps xl for residual)
  ln_kernel<<<TOK / 8, 256, 0, stream>>>(x, ln1_g, ln1_b, xl, xn, 1);

  // QKV GEMM: [4096,256] x [768,256]^T -> f16 [4096,768]
  gemm_wmma_kernel<<<dim3(768 / 64, TOK / 128), 256, 0, stream>>>(
      xn, wqkv16, b_qkv, nullptr, qkv16, TOK, 768, 256, 0);

  // flash attention -> att16 [4096,256] f16
  attn_kernel<<<dim3(TOK / (16 * 8), HEADS), 256, 0, stream>>>(qkv16, att16);

  // proj GEMM + residual: xl2 = xl + att*Wp^T + bp  (f32)
  gemm_wmma_kernel<<<dim3(256 / 64, TOK / 128), 256, 0, stream>>>(
      att16, wproj16, b_proj, xl, xl2, TOK, 256, 256, 1);

  // LN2
  ln_kernel<<<TOK / 8, 256, 0, stream>>>(xl2, ln2_g, ln2_b, nullptr, xn2, 0);

  // MLP1 GEMM + exact GELU -> h1 f16 [4096,1024]
  gemm_wmma_kernel<<<dim3(1024 / 64, TOK / 128), 256, 0, stream>>>(
      xn2, wmlp116, b_mlp1, nullptr, h1, TOK, 1024, 256, 2);

  // MLP2 GEMM + residual + transpose store to NCHW output
  gemm_wmma_kernel<<<dim3(256 / 64, TOK / 128), 256, 0, stream>>>(
      h1, wmlp216, b_mlp2, xl2, d_out, TOK, 256, 1024, 3);
}
